// MultiHeadSelfAttention_55903294325061
// MI455X (gfx1250) — matrix operand order repaired and checked by reading, not compiled, not run on hardware
//
#include <hip/hip_runtime.h>

// ---------------------------------------------------------------------------
// MHA forward for MI455X (gfx1250). bf16 WMMA, f32 accumulate, TDM-fed LDS.
//   B=4, S=2048, D=1024, H=16, Dh=64
// Pipeline:
//   cvt_bf16 x3, cvt_wt (transpose+convert) x4      -> bf16 operands in ws
//   gemm<headsplit> x3 (TDM double-buffered tiles)  -> Qp/Kp/Vp [B,H,S,64]
//   flash-attention (TDM K tile, DPP softmax)       -> Ao [B,S,1024] bf16
//   gemm<f32-out>                                   -> d_out [B,S,1024] f32
// Workspace: 3*16MiB (qkv bf16) + 4*2MiB (W^T bf16) + 3*16MiB (proj)
//            + 16MiB (Ao) = 120 MiB.
// ---------------------------------------------------------------------------

#define DMODEL 1024
#define NHEADS 16
#define HDIM   64
#define BATCH  4
#define SEQ    2048
#define MROWS  (BATCH * SEQ)   // 8192

typedef __bf16 bf16_t;
typedef bf16_t v16bf __attribute__((ext_vector_type(16)));
typedef float  v8f   __attribute__((ext_vector_type(8)));
typedef unsigned int u32x4 __attribute__((ext_vector_type(4)));
typedef int          i32x8 __attribute__((ext_vector_type(8)));
typedef int          i32x4 __attribute__((ext_vector_type(4)));

union Frag {
  v16bf v;
  unsigned short u[16];
};

__device__ __forceinline__ unsigned short f2bf(float f) {
  union { float f; unsigned int u; } x; x.f = f;
  unsigned int r = x.u + 0x7FFFu + ((x.u >> 16) & 1u);  // round-to-nearest-even
  return (unsigned short)(r >> 16);
}

__device__ __forceinline__ v8f v8f_zero() {
  v8f z = {0.f, 0.f, 0.f, 0.f, 0.f, 0.f, 0.f, 0.f};
  return z;
}

__device__ __forceinline__ v8f wmma_bf16(v16bf a, v16bf b, v8f c) {
  return __builtin_amdgcn_wmma_f32_16x16x32_bf16(false, a, false, b, (short)0, c,
                                                 false, false);
}

// ---- 16-lane-group reductions via DPP (no LDS permute traffic) -------------
//  quad_perm[1,0,3,2]=0xB1 (xor1), quad_perm[2,3,0,1]=0x4E (xor2),
//  row_ror:4=0x124, row_ror:8=0x128 complete the 16-lane row.
template <int CTRL>
__device__ __forceinline__ float dpp_movf(float x) {
  return __int_as_float(
      __builtin_amdgcn_update_dpp(0, __float_as_int(x), CTRL, 0xF, 0xF, true));
}
__device__ __forceinline__ float red16_max(float x) {
  x = fmaxf(x, dpp_movf<0xB1>(x));
  x = fmaxf(x, dpp_movf<0x4E>(x));
  x = fmaxf(x, dpp_movf<0x124>(x));
  x = fmaxf(x, dpp_movf<0x128>(x));
  return x;
}
__device__ __forceinline__ float red16_sum(float x) {
  x += dpp_movf<0xB1>(x);
  x += dpp_movf<0x4E>(x);
  x += dpp_movf<0x124>(x);
  x += dpp_movf<0x128>(x);
  return x;
}

// ---- Tensor Data Mover: 2D bf16 tile -> padded LDS (ISA 08 §8 D# layout) ---
//  pad_interval_code: rows of (2<<code) dwords; pad_amount_code: (code+1) dwords.
//  clang-23 / therock-10.0 builtin form: (g0, g1, g2, g3, g4, cpol).
__device__ __forceinline__ void tdm_load_tile_bf16(
    unsigned lds_byte_addr, const unsigned short* gptr, unsigned tile_d0,
    unsigned tile_d1, unsigned long long row_stride_elems,
    unsigned pad_interval_code, unsigned pad_amount_code) {
  const unsigned long long ga = (unsigned long long)gptr;
  u32x4 g0;
  g0[0] = 1u;                                                  // count=1
  g0[1] = lds_byte_addr;                                       // lds_addr
  g0[2] = (unsigned)ga;                                        // global[31:0]
  g0[3] = (unsigned)((ga >> 32) & 0x01FFFFFFull) | (2u << 30); // global[56:32]|type=2
  i32x8 g1;
  g1[0] = (int)((1u << 16) | (1u << 20) |                      // data_size=2B, pad_en
                (pad_interval_code << 22) | (pad_amount_code << 25));
  g1[1] = (int)((tile_d0 & 0xFFFFu) << 16);                    // tensor_dim0[15:0]
  g1[2] = (int)(((tile_d0 >> 16) & 0xFFFFu) | ((tile_d1 & 0xFFFFu) << 16));
  g1[3] = (int)(((tile_d1 >> 16) & 0xFFFFu) | (tile_d0 << 16)); // tile_dim0
  g1[4] = (int)(tile_d1 & 0xFFFFu);                            // tile_dim1, tile_dim2=0
  g1[5] = (int)(row_stride_elems & 0xFFFFFFFFull);             // dim0_stride[31:0]
  g1[6] = (int)((row_stride_elems >> 32) & 0xFFFFull);         // dim0_stride[47:32]
  g1[7] = 0;
  i32x4 z4 = {0, 0, 0, 0};
  i32x8 z8 = {0, 0, 0, 0, 0, 0, 0, 0};
  __builtin_amdgcn_tensor_load_to_lds(g0, g1, z4, z4, z8, 0);
}

__device__ __forceinline__ unsigned lds_off(const void* p) {
  // flat LDS aperture occupies a 4GB-aligned window: low 32 bits == LDS address
  return (unsigned)(unsigned long long)p;
}

// ---------------------------------------------------------------------------
// One-time converts: f32 -> bf16 (flat), and W f32 -> W^T bf16 (tiled).
// ---------------------------------------------------------------------------
__global__ void __launch_bounds__(256)
cvt_bf16_kernel(const float* __restrict__ in, unsigned short* __restrict__ out,
                int n) {
  const int i = (blockIdx.x * 256 + threadIdx.x) * 8;
  if (i + 8 > n) return;
  const float4 a = ((const float4*)(in + i))[0];
  const float4 b = ((const float4*)(in + i))[1];
  uint4 o;
  o.x = (unsigned)f2bf(a.x) | ((unsigned)f2bf(a.y) << 16);
  o.y = (unsigned)f2bf(a.z) | ((unsigned)f2bf(a.w) << 16);
  o.z = (unsigned)f2bf(b.x) | ((unsigned)f2bf(b.y) << 16);
  o.w = (unsigned)f2bf(b.z) | ((unsigned)f2bf(b.w) << 16);
  ((uint4*)(out + i))[0] = o;
}

__global__ void __launch_bounds__(256)
cvt_wt_kernel(const float* __restrict__ W, unsigned short* __restrict__ WT) {
  __shared__ unsigned short t[32][33];
  const int bx = blockIdx.x * 32, by = blockIdx.y * 32;   // bx: n-tile, by: k-tile
  const int x = threadIdx.x & 31, y = threadIdx.x >> 5;   // y: 0..7
#pragma unroll
  for (int i = 0; i < 4; ++i) {
    const int r = y + i * 8;                              // k_local
    t[r][x] = f2bf(W[(size_t)(by + r) * DMODEL + bx + x]);
  }
  __syncthreads();
#pragma unroll
  for (int i = 0; i < 4; ++i) {
    const int r = y + i * 8;                              // n_local
    WT[(size_t)(bx + r) * DMODEL + by + x] = t[x][r];     // WT[n][k] = W[k][n]
  }
}

// ---------------------------------------------------------------------------
// GEMM: Out[8192,1024] = A[8192,1024]_bf16 x BT[1024(n),1024(k)]_bf16^T + bias
//  Block tile 128x128, 8 waves (4x2) of 32x64 subtiles (8 WMMA accumulators).
//  Both tiles DMA'd by TDM into pad-40 LDS rows, double-buffered; waves only
//  do ds_load_b128 fragment reads + v_wmma.
//  OMODE: 0 = bf16 head-split [B,H,S,64]; 2 = f32 flat [M,N].
// ---------------------------------------------------------------------------
template <int OMODE>
__global__ void __launch_bounds__(256)
mha_gemm_kernel(const unsigned short* __restrict__ A,
                const unsigned short* __restrict__ BT,
                const float* __restrict__ bias, void* __restrict__ Out) {
  constexpr int BM = 128, BN = 128, BK = 32;
  constexpr int LDA = BK + 8;  // 40 ushorts = 80B rows (16B aligned, conflict-free)
  __shared__ unsigned short As[2][BM * LDA];
  __shared__ unsigned short Bs[2][BN * LDA];

  const int tid  = threadIdx.x;
  const int lane = tid & 31;
  const int wave = tid >> 5;
  const int wm   = wave >> 1;
  const int wn   = wave & 1;
  const int bm   = blockIdx.x * BM;
  const int bn   = blockIdx.y * BN;

  v8f acc[2][4];
#pragma unroll
  for (int i = 0; i < 2; ++i)
#pragma unroll
    for (int j = 0; j < 4; ++j) acc[i][j] = v8f_zero();

  const int m    = lane & 15;
  const int hh   = (lane >> 4) * 8;
  const int nl   = lane & 15;
  const int kb2  = (lane >> 4) * 16;
  const int half = lane >> 4;

  constexpr int NIT = DMODEL / BK;  // 32
  if (wave == 0) {  // prime buffer 0 (rows = 16 dwords -> code 3; pad 4 dw -> code 3)
    tdm_load_tile_bf16(lds_off(&As[0][0]), A + (size_t)bm * DMODEL, BK, BM, DMODEL, 3, 3);
    tdm_load_tile_bf16(lds_off(&Bs[0][0]), BT + (size_t)bn * DMODEL, BK, BN, DMODEL, 3, 3);
  }

  for (int it = 0; it < NIT; ++it) {
    const int cur = it & 1;
    if (wave == 0) {
      if (it + 1 < NIT) {
        const int k1 = (it + 1) * BK;
        tdm_load_tile_bf16(lds_off(&As[cur ^ 1][0]), A + (size_t)bm * DMODEL + k1,
                           BK, BM, DMODEL, 3, 3);
        tdm_load_tile_bf16(lds_off(&Bs[cur ^ 1][0]), BT + (size_t)bn * DMODEL + k1,
                           BK, BN, DMODEL, 3, 3);
        __builtin_amdgcn_s_wait_tensorcnt(2);  // current pair landed (in-order)
      } else {
        __builtin_amdgcn_s_wait_tensorcnt(0);
      }
    }
    __syncthreads();  // publish current tiles

    Frag af[2], bf[4];
#pragma unroll
    for (int tm = 0; tm < 2; ++tm) {
      const unsigned short* p = &As[cur][(wm * 32 + tm * 16 + m) * LDA];
#pragma unroll
      for (int j = 0; j < 16; ++j) af[tm].u[j] = p[kb2 + j];
    }
#pragma unroll
    for (int tn = 0; tn < 4; ++tn) {
      const unsigned short* p = &Bs[cur][(wn * 64 + tn * 16 + nl) * LDA + kb2];
#pragma unroll
      for (int j = 0; j < 16; ++j) bf[tn].u[j] = p[j];
    }
#pragma unroll
    for (int tm = 0; tm < 2; ++tm)
#pragma unroll
      for (int tn = 0; tn < 4; ++tn)
        acc[tm][tn] = wmma_bf16(af[tm].v, bf[tn].v, acc[tm][tn]);
    __syncthreads();  // consumed: safe to DMA into this buffer next round
  }

#pragma unroll
  for (int tm = 0; tm < 2; ++tm) {
#pragma unroll
    for (int tn = 0; tn < 4; ++tn) {
      const int gn = bn + wn * 64 + tn * 16 + nl;
      const float bv = bias[gn];
#pragma unroll
      for (int r = 0; r < 8; ++r) {
        const int gm = bm + wm * 32 + tm * 16 + r + 8 * half;
        const float val = acc[tm][tn][r] + bv;
        if (OMODE == 0) {
          const int b = gm >> 11, s = gm & (SEQ - 1);
          const int h = gn >> 6, d = gn & (HDIM - 1);
          ((unsigned short*)Out)[(((size_t)b * NHEADS + h) * SEQ + s) * HDIM + d] =
              f2bf(val);
        } else {
          ((float*)Out)[(size_t)gm * DMODEL + gn] = val;
        }
      }
    }
  }
}

// ---------------------------------------------------------------------------
// Flash attention: one workgroup per (b,h,128 q-rows); 8 waves x 16 q-rows.
//  K tile DMA'd by TDM (padded rows); V transposed manually; online softmax
//  row stats via DPP 16-lane reductions (C-frag rows == VGPR index + lane half).
// ---------------------------------------------------------------------------
__global__ void __launch_bounds__(256)
mha_attn_kernel(const unsigned short* __restrict__ Q,
                const unsigned short* __restrict__ K,
                const unsigned short* __restrict__ V,
                const float* __restrict__ mask,
                unsigned short* __restrict__ AO) {
  constexpr int QB = 128, KT = 32;
  constexpr int LK = HDIM + 8;  // 72 (TDM pad: 32-dw rows + 4 dw)
  constexpr int LV = KT + 8;    // 40
  constexpr int LP = KT + 8;    // 40
  __shared__ unsigned short Kt[KT * LK];
  __shared__ unsigned short Vtt[HDIM * LV];
  __shared__ unsigned short Ps[8 * 16 * LP];

  const int bh   = blockIdx.x;
  const int b    = bh >> 4;
  const int h    = bh & 15;
  const int q0   = blockIdx.y * QB;
  const int tid  = threadIdx.x;
  const int lane = tid & 31;
  const int wave = tid >> 5;

  const unsigned short* Qh = Q + (size_t)bh * SEQ * HDIM;
  const unsigned short* Kh = K + (size_t)bh * SEQ * HDIM;
  const unsigned short* Vh = V + (size_t)bh * SEQ * HDIM;
  const float* mk = mask + (size_t)b * SEQ;

  const int m    = lane & 15;
  const int hh   = (lane >> 4) * 8;
  const int nl   = lane & 15;
  const int kb2  = (lane >> 4) * 16;
  const int half = lane >> 4;

  Frag qf[2];
  {
    const unsigned short* qp = Qh + (size_t)(q0 + wave * 16 + m) * HDIM;
#pragma unroll
    for (int c = 0; c < 2; ++c)
#pragma unroll
      for (int j = 0; j < 16; ++j) qf[c].u[j] = qp[c * 32 + kb2 + j];
  }

  v8f o[4];
#pragma unroll
  for (int t = 0; t < 4; ++t) o[t] = v8f_zero();
  float mi[8], li[8];
#pragma unroll
  for (int r = 0; r < 8; ++r) { mi[r] = -1e30f; li[r] = 0.f; }

  unsigned short* Pw = &Ps[wave * 16 * LP];
  const float scale = 0.125f;

  for (int k0 = 0; k0 < SEQ; k0 += KT) {
    // ---- K tile via TDM (rows = 32 dwords -> code 4; pad 4 dw -> code 3) ----
    if (wave == 0)
      tdm_load_tile_bf16(lds_off(&Kt[0]), Kh + (size_t)k0 * HDIM, HDIM, KT, HDIM, 4, 3);
    // ---- V tile transposed manually by all 256 threads ----
    const unsigned int* vg = (const unsigned int*)(Vh + (size_t)k0 * HDIM);
#pragma unroll
    for (int i = tid; i < KT * HDIM / 2; i += 256) {
      const int key = i >> 5, dp = i & 31;
      const unsigned int vv = vg[i];
      Vtt[(dp * 2)     * LV + key] = (unsigned short)(vv & 0xffffu);
      Vtt[(dp * 2 + 1) * LV + key] = (unsigned short)(vv >> 16);
    }
    if (k0 + KT < SEQ)
      __builtin_prefetch(Vh + (size_t)(k0 + KT) * HDIM + lane * 64, 0, 0);
    if (wave == 0) __builtin_amdgcn_s_wait_tensorcnt(0);
    __syncthreads();

    // ---- logits: S[16 q x 32 keys] ----
    v8f s0 = v8f_zero(), s1 = v8f_zero();
#pragma unroll
    for (int c = 0; c < 2; ++c) {
      Frag kf0, kf1;
      const unsigned short* kp0 = &Kt[nl * LK + c * 32 + kb2];
      const unsigned short* kp1 = &Kt[(16 + nl) * LK + c * 32 + kb2];
#pragma unroll
      for (int j = 0; j < 16; ++j) { kf0.u[j] = kp0[j]; kf1.u[j] = kp1[j]; }
      s0 = wmma_bf16(qf[c].v, kf0.v, s0);
      s1 = wmma_bf16(qf[c].v, kf1.v, s1);
    }

    // ---- online softmax (DPP reductions, no LDS permutes) ----
    const float mk0 = mk[k0 + nl] * -1e9f;
    const float mk1 = mk[k0 + 16 + nl] * -1e9f;
    float alpha[8];
#pragma unroll
    for (int r = 0; r < 8; ++r) {
      const float x0 = s0[r] * scale + mk0;
      const float x1 = s1[r] * scale + mk1;
      const float t    = red16_max(fmaxf(x0, x1));
      const float mnew = fmaxf(mi[r], t);
      const float a  = __expf(mi[r] - mnew);
      const float p0 = __expf(x0 - mnew);
      const float p1 = __expf(x1 - mnew);
      const float rs = red16_sum(p0 + p1);
      li[r] = li[r] * a + rs;
      mi[r] = mnew;
      alpha[r] = a;
      Pw[(r + 8 * half) * LP + nl]      = f2bf(p0);
      Pw[(r + 8 * half) * LP + 16 + nl] = f2bf(p1);
    }

#pragma unroll
    for (int t = 0; t < 4; ++t)
#pragma unroll
      for (int r = 0; r < 8; ++r) o[t][r] = o[t][r] * alpha[r];

    // ---- O += P x V ----
    Frag pf;
    {
      const unsigned short* pp = &Pw[m * LP];
#pragma unroll
      for (int j = 0; j < 16; ++j) pf.u[j] = pp[kb2 + j];
    }
#pragma unroll
    for (int t = 0; t < 4; ++t) {
      Frag vf;
      const unsigned short* vp = &Vtt[(t * 16 + nl) * LV + kb2];
#pragma unroll
      for (int j = 0; j < 16; ++j) vf.u[j] = vp[j];
      o[t] = wmma_bf16(pf.v, vf.v, o[t]);
    }
    __syncthreads();
  }

#pragma unroll
  for (int t = 0; t < 4; ++t) {
#pragma unroll
    for (int r = 0; r < 8; ++r) {
      const int s = q0 + wave * 16 + r + 8 * half;
      const int d = t * 16 + nl;
      const float val = o[t][r] / li[r];
      AO[((size_t)b * SEQ + s) * DMODEL + h * HDIM + d] = f2bf(val);
    }
  }
}

// ---------------------------------------------------------------------------
extern "C" void kernel_launch(void* const* d_in, const int* in_sizes, int n_in,
                              void* d_out, int out_size, void* d_ws, size_t ws_size,
                              hipStream_t stream) {
  (void)in_sizes; (void)n_in; (void)out_size; (void)ws_size;
  const float* q    = (const float*)d_in[0];
  const float* k    = (const float*)d_in[1];
  const float* v    = (const float*)d_in[2];
  const float* mask = (const float*)d_in[3];
  const float* Wq   = (const float*)d_in[4];
  const float* bq   = (const float*)d_in[5];
  const float* Wk   = (const float*)d_in[6];
  const float* bk   = (const float*)d_in[7];
  const float* Wv   = (const float*)d_in[8];
  const float* bv   = (const float*)d_in[9];
  const float* Wo   = (const float*)d_in[10];
  const float* bo   = (const float*)d_in[11];

  const size_t PROJ = (size_t)MROWS * DMODEL;   // 8M elems
  const size_t WSZ  = (size_t)DMODEL * DMODEL;  // 1M elems
  unsigned short* qb  = (unsigned short*)d_ws;
  unsigned short* kb  = qb + PROJ;
  unsigned short* vb  = kb + PROJ;
  unsigned short* WqT = vb + PROJ;
  unsigned short* WkT = WqT + WSZ;
  unsigned short* WvT = WkT + WSZ;
  unsigned short* WoT = WvT + WSZ;
  unsigned short* Qp  = WoT + WSZ;
  unsigned short* Kp  = Qp + PROJ;
  unsigned short* Vp  = Kp + PROJ;
  unsigned short* Ao  = Vp + PROJ;

  const int cvtBlocks = (int)(PROJ / (256 * 8));
  cvt_bf16_kernel<<<cvtBlocks, 256, 0, stream>>>(q, qb, (int)PROJ);
  cvt_bf16_kernel<<<cvtBlocks, 256, 0, stream>>>(k, kb, (int)PROJ);
  cvt_bf16_kernel<<<cvtBlocks, 256, 0, stream>>>(v, vb, (int)PROJ);
  dim3 wt(DMODEL / 32, DMODEL / 32, 1);
  cvt_wt_kernel<<<wt, 256, 0, stream>>>(Wq, WqT);
  cvt_wt_kernel<<<wt, 256, 0, stream>>>(Wk, WkT);
  cvt_wt_kernel<<<wt, 256, 0, stream>>>(Wv, WvT);
  cvt_wt_kernel<<<wt, 256, 0, stream>>>(Wo, WoT);

  dim3 gg(MROWS / 128, DMODEL / 128, 1);
  mha_gemm_kernel<0><<<gg, 256, 0, stream>>>(qb, WqT, bq, Qp);
  mha_gemm_kernel<0><<<gg, 256, 0, stream>>>(kb, WkT, bk, Kp);
  mha_gemm_kernel<0><<<gg, 256, 0, stream>>>(vb, WvT, bv, Vp);
  mha_attn_kernel<<<dim3(BATCH * NHEADS, SEQ / 128, 1), 256, 0, stream>>>(
      Qp, Kp, Vp, mask, Ao);
  mha_gemm_kernel<2><<<gg, 256, 0, stream>>>(Ao, WoT, bo, d_out);
}
